// PointNet2Regressor_StrokeMasks_15324443312194
// MI455X (gfx1250) — compile-verified, matrix operand order checked by reading, not run on hardware
//
#include <hip/hip_runtime.h>
#include <math.h>

// ---------------- CDNA5 WMMA types ----------------
typedef __attribute__((ext_vector_type(16))) _Float16 v16h;
typedef __attribute__((ext_vector_type(8)))  float    v8f;

#define B_    16
#define N_    4096
#define NP1   512
#define NS1   32
#define NP2   128
#define NS2   64
#define OUTV  1500
#define NPADH 4512   // 4500 rounded up to 16

// =====================================================================
// (B,3,N) -> (B,N,3)
__global__ void k_xyzT(const float* __restrict__ xyz, float* __restrict__ pts) {
  int t = blockIdx.x * blockDim.x + threadIdx.x;
  if (t >= B_ * N_) return;
  int b = t / N_, i = t % N_;
#pragma unroll
  for (int c = 0; c < 3; ++c)
    pts[(size_t)t * 3 + c] = xyz[((size_t)b * 3 + c) * N_ + i];
}

// =====================================================================
// Farthest point sampling: one 256-thread block per batch.
// Matches jax.lax.scan: idx[0]=0; each step outputs 'far' then updates.
__global__ void k_fps(const float* __restrict__ pts, int n, int npoint,
                      int* __restrict__ idxOut) {
  int b = blockIdx.x;
  const float* p = pts + (size_t)b * n * 3;
  __shared__ float dist[N_];
  __shared__ float rbest[256];
  __shared__ int   ribest[256];
  int tid = threadIdx.x;
  for (int i = tid; i < n; i += 256) dist[i] = 1e10f;
  __syncthreads();
  int far = 0;
  for (int s = 0; s < npoint; ++s) {
    if (tid == 0) idxOut[b * npoint + s] = far;
    float fx = p[far * 3 + 0], fy = p[far * 3 + 1], fz = p[far * 3 + 2];
    float best = -1.0f; int bi = n;
    for (int i = tid; i < n; i += 256) {
      float dx = p[i * 3 + 0] - fx, dy = p[i * 3 + 1] - fy, dz = p[i * 3 + 2] - fz;
      float d = dx * dx + dy * dy + dz * dz;
      float nd = fminf(dist[i], d);
      dist[i] = nd;
      if (nd > best) { best = nd; bi = i; }   // strict > keeps lowest index in-thread
    }
    rbest[tid] = best; ribest[tid] = bi;
    __syncthreads();
    for (int off = 128; off > 0; off >>= 1) {
      if (tid < off) {
        float ob = rbest[tid + off]; int oi = ribest[tid + off];
        if (ob > rbest[tid] || (ob == rbest[tid] && oi < ribest[tid])) {
          rbest[tid] = ob; ribest[tid] = oi;   // tie-break: first (lowest) index
        }
      }
      __syncthreads();
    }
    far = ribest[0];
    __syncthreads();
  }
}

// =====================================================================
__global__ void k_gather_ctr(const float* __restrict__ pts, const int* __restrict__ idx,
                             int n, int S, float* __restrict__ ctr) {
  int t = blockIdx.x * blockDim.x + threadIdx.x;
  if (t >= B_ * S) return;
  int b = t / S;
  int i = idx[t];
#pragma unroll
  for (int c = 0; c < 3; ++c)
    ctr[(size_t)t * 3 + c] = pts[((size_t)b * n + i) * 3 + c];
}

// =====================================================================
// Ball query: one wave32 per center; ascending-index in-ball selection via
// ballot + prefix popcount, padded with the first in-ball index (the center
// itself is always in-ball since centers are sampled from pts).
__global__ void k_ballq(const float* __restrict__ pts, const float* __restrict__ ctr,
                        int n, int S, int nsample, float r2, int* __restrict__ out) {
  int bs = blockIdx.x;             // b*S + s
  int b = bs / S;
  int lane = threadIdx.x;          // 0..31 (wave32)
  const float* p = pts + (size_t)b * n * 3;
  float cx = ctr[bs * 3 + 0], cy = ctr[bs * 3 + 1], cz = ctr[bs * 3 + 2];
  int* o = out + (size_t)bs * nsample;
  int cnt = 0, first = -1;
  for (int base = 0; base < n && cnt < nsample; base += 32) {
    int i = base + lane;
    bool inb = false;
    if (i < n) {
      float dx = p[i * 3 + 0] - cx, dy = p[i * 3 + 1] - cy, dz = p[i * 3 + 2] - cz;
      inb = (dx * dx + dy * dy + dz * dz) <= r2;
    }
    unsigned mask = (unsigned)__ballot(inb);
    int rank = __popc(mask & ((1u << lane) - 1u));
    if (inb) {
      int pos = cnt + rank;
      if (pos < nsample) o[pos] = i;
    }
    if (first < 0 && mask) first = base + (__ffs(mask) - 1);
    cnt += __popc(mask);
  }
  if (cnt < nsample) {
    for (int j = cnt + lane; j < nsample; j += 32) o[j] = first;
  }
}

// =====================================================================
// Grouped-input builders (f16, zero-padded K dimension).
__global__ void k_group1(const float* __restrict__ pts, const float* __restrict__ ctr,
                         const int* __restrict__ bq, _Float16* __restrict__ X) {
  int t = blockIdx.x * blockDim.x + threadIdx.x;           // (b,s,k)
  if (t >= B_ * NP1 * NS1) return;
  int bs = t / NS1, b = bs / NP1;
  int i = bq[t];
  _Float16* row = X + (size_t)t * 32;
  const float* pp = pts + ((size_t)b * N_ + i) * 3;
  const float* cc = ctr + (size_t)bs * 3;
  row[0] = (_Float16)(pp[0] - cc[0]);
  row[1] = (_Float16)(pp[1] - cc[1]);
  row[2] = (_Float16)(pp[2] - cc[2]);
  for (int c = 3; c < 32; ++c) row[c] = (_Float16)0.0f;
}

__global__ void k_group2(const float* __restrict__ pts, const float* __restrict__ ctr,
                         const int* __restrict__ bq, const _Float16* __restrict__ feat,
                         _Float16* __restrict__ X) {
  int t = blockIdx.x * blockDim.x + threadIdx.x;           // (b,s,k)
  if (t >= B_ * NP2 * NS2) return;
  int bs = t / NS2, b = bs / NP2;
  int i = bq[t];                                           // index into 512 sa1 centers
  _Float16* row = X + (size_t)t * 160;
  const float* pp = pts + ((size_t)b * NP1 + i) * 3;
  const float* cc = ctr + (size_t)bs * 3;
  row[0] = (_Float16)(pp[0] - cc[0]);
  row[1] = (_Float16)(pp[1] - cc[1]);
  row[2] = (_Float16)(pp[2] - cc[2]);
  const _Float16* f = feat + ((size_t)b * NP1 + i) * 128;
  for (int c = 0; c < 128; ++c) row[3 + c] = f[c];
  for (int c = 131; c < 160; ++c) row[c] = (_Float16)0.0f;
}

__global__ void k_group3(const float* __restrict__ ctr2, const _Float16* __restrict__ feat,
                         _Float16* __restrict__ X) {
  int t = blockIdx.x * blockDim.x + threadIdx.x;           // (b, j<128)
  if (t >= B_ * NP2) return;
  _Float16* row = X + (size_t)t * 288;
  row[0] = (_Float16)ctr2[t * 3 + 0];
  row[1] = (_Float16)ctr2[t * 3 + 1];
  row[2] = (_Float16)ctr2[t * 3 + 2];
  const _Float16* f = feat + (size_t)t * 256;
  for (int c = 0; c < 256; ++c) row[3 + c] = f[c];
  for (int c = 259; c < 288; ++c) row[c] = (_Float16)0.0f;
}

// =====================================================================
// Weight prep: f32 (N,Kin) -> f16 (Npad,Kpad) zero-padded; fold eval-BN:
//   y = relu((xW + b)*s + be),  s = g/sqrt(1+eps)  =>  scale=s, shift=b*s+be
__global__ void k_prep(const float* __restrict__ W, const float* __restrict__ b,
                       const float* __restrict__ g, const float* __restrict__ be,
                       int N, int Kin, int Kpad, int Npad,
                       _Float16* __restrict__ w16, float* __restrict__ sc,
                       float* __restrict__ sh) {
  int t = blockIdx.x * blockDim.x + threadIdx.x;
  if (t >= Npad * Kpad) return;
  int n = t / Kpad, k = t % Kpad;
  float v = (n < N && k < Kin) ? W[(size_t)n * Kin + k] : 0.0f;
  w16[t] = (_Float16)v;
  if (k == 0) {
    if (n < N) {
      float inv = rsqrtf(1.0f + 1e-5f);
      float s = g ? inv * g[n] : 1.0f;
      sc[n] = s;
      sh[n] = b[n] * s + (be ? be[n] : 0.0f);
    } else {
      sc[n] = 0.0f; sh[n] = 0.0f;
    }
  }
}

// =====================================================================
// WMMA GEMM: one wave (32 threads) per 16x16 output tile.
//   Y[m][n] = act( (sum_k X[m][k]*W[n][k]) * sc[n] + sh[n] )
// A fragment: lane (0..15) holds row m=lane, K=kb..kb+15 contiguous f16;
//             lane+16 holds same row, K=kb+16.. (contiguous K order, B uses the same).
// B fragment: lane holds W-row n=lane&15, same contiguous K slice (B = W^T).
// D layout:   VGPR i -> row = i + 8*(lane>>4), col = lane&15.
template <int ACT, bool OUT16>
__global__ void k_gemm(const _Float16* __restrict__ X, const _Float16* __restrict__ Wt,
                       int K,                      // padded K (= ldx = ldw), mult of 32
                       const float* __restrict__ sc, const float* __restrict__ sh,
                       _Float16* __restrict__ Y16, float* __restrict__ Y32,
                       int ldy, int Nreal) {
  int tileM = blockIdx.x, tileN = blockIdx.y;
  int lane = threadIdx.x;                          // 0..31 wave32
  int rA = tileM * 16 + (lane & 15);
  int rW = tileN * 16 + (lane & 15);
  int kb = (lane >> 4) * 16;
  const _Float16* xp = X + (size_t)rA * K + kb;
  const _Float16* wp = Wt + (size_t)rW * K + kb;
  v8f acc = {};
  for (int k0 = 0; k0 < K; k0 += 32) {
    __builtin_prefetch(xp + k0 + 128, 0, 0);       // global_prefetch_b8
    v16h a = *(const v16h*)(xp + k0);
    v16h bfrag = *(const v16h*)(wp + k0);
    acc = __builtin_amdgcn_wmma_f32_16x16x32_f16(
        /*neg_a=*/false, a, /*neg_b=*/false, bfrag,
        /*c_mod=*/(short)0, acc, /*reuse_a=*/false, /*reuse_b=*/false);
  }
  int col = tileN * 16 + (lane & 15);
  int rowh = 8 * (lane >> 4);
  float s = sc[col], t = sh[col];
#pragma unroll
  for (int i = 0; i < 8; ++i) {
    int row = tileM * 16 + rowh + i;
    float v = acc[i] * s + t;
    if (ACT == 1) v = fmaxf(v, 0.0f);
    if (col < Nreal) {
      if (OUT16) Y16[(size_t)row * ldy + col] = (_Float16)v;
      else       Y32[(size_t)row * ldy + col] = v;
    }
  }
}

// =====================================================================
// Max over K neighbor rows (post-ReLU features, so result >= 0 like ref).
__global__ void k_maxpool(const _Float16* __restrict__ in, _Float16* __restrict__ out,
                          int G, int K, int C) {
  int t = blockIdx.x * blockDim.x + threadIdx.x;
  if (t >= G * C) return;
  int g = t / C, c = t % C;
  const _Float16* p = in + ((size_t)g * K) * C + c;
  float m = -1e30f;
  for (int k = 0; k < K; ++k) m = fmaxf(m, (float)p[(size_t)k * C]);
  out[t] = (_Float16)m;
}

// =====================================================================
// coords pass-through; normals: tanh then L2-normalize; interleave (B,1500,6).
__global__ void k_final(const float* __restrict__ co, const float* __restrict__ nr,
                        float* __restrict__ out) {
  int t = blockIdx.x * blockDim.x + threadIdx.x;
  if (t >= B_ * OUTV) return;
  int b = t / OUTV, v = t % OUTV;
  const float* c = co + (size_t)b * NPADH + v * 3;
  const float* n = nr + (size_t)b * NPADH + v * 3;
  float nx = tanhf(n[0]), ny = tanhf(n[1]), nz = tanhf(n[2]);
  float len = fmaxf(sqrtf(nx * nx + ny * ny + nz * nz), 1e-12f);
  float* o = out + (size_t)t * 6;
  o[0] = c[0]; o[1] = c[1]; o[2] = c[2];
  o[3] = nx / len; o[4] = ny / len; o[5] = nz / len;
}

// =====================================================================
struct LayerDef { int N, Kin, Kpad, Npad, iW, ib, ig, ibe; };

extern "C" void kernel_launch(void* const* d_in, const int* in_sizes, int n_in,
                              void* d_out, int out_size, void* d_ws, size_t ws_size,
                              hipStream_t stream) {
  (void)in_sizes; (void)n_in; (void)out_size; (void)ws_size;
  // Input order: xyz first (setup_inputs dict order), then params flattened
  // jax-pytree style (dict keys sorted): bn1{be,g} bn2{be,g} fc1{W,b} fc2{W,b}
  // fc3{W,b} fcn{W,b} sa1[0..2]{W,b,be,g} sa2[...] sa3[...]
  const float* xyz = (const float*)d_in[0];
  static const LayerDef L[13] = {
      {  64,    3,   32,   64, 13, 14, 16, 15},   // sa1.0
      {  64,   64,   64,   64, 17, 18, 20, 19},   // sa1.1
      { 128,   64,   64,  128, 21, 22, 24, 23},   // sa1.2
      { 128,  131,  160,  128, 25, 26, 28, 27},   // sa2.0
      { 128,  128,  128,  128, 29, 30, 32, 31},   // sa2.1
      { 256,  128,  128,  256, 33, 34, 36, 35},   // sa2.2
      { 256,  259,  288,  256, 37, 38, 40, 39},   // sa3.0
      { 512,  256,  256,  512, 41, 42, 44, 43},   // sa3.1
      {1024,  512,  512, 1024, 45, 46, 48, 47},   // sa3.2
      {1024, 1024, 1024, 1024,  5,  6,  2,  1},   // fc1 (+bn1)
      {1024, 1024, 1024, 1024,  7,  8,  4,  3},   // fc2 (+bn2)
      {4500, 1024, 1024, NPADH, 9, 10, -1, -1},   // fc3 (no BN)
      {4500, 1024, 1024, NPADH, 11, 12, -1, -1},  // fcn (no BN)
  };

  // ---- workspace carve-out (~158 MB) ----
  char* ws = (char*)d_ws;
  size_t cur = 0;
  auto alloc = [&](size_t bytes) -> void* {
    void* p = ws + cur;
    cur += (bytes + 255) & ~(size_t)255;
    return p;
  };
  float*     ptsT  = (float*)alloc((size_t)B_ * N_ * 3 * 4);
  int*       idx1  = (int*)alloc((size_t)B_ * NP1 * 4);
  float*     ctr1  = (float*)alloc((size_t)B_ * NP1 * 3 * 4);
  int*       bq1   = (int*)alloc((size_t)B_ * NP1 * NS1 * 4);
  int*       idx2  = (int*)alloc((size_t)B_ * NP2 * 4);
  float*     ctr2  = (float*)alloc((size_t)B_ * NP2 * 3 * 4);
  int*       bq2   = (int*)alloc((size_t)B_ * NP2 * NS2 * 4);
  _Float16*  l1f   = (_Float16*)alloc((size_t)B_ * NP1 * 128 * 2);
  _Float16*  l2f   = (_Float16*)alloc((size_t)B_ * NP2 * 256 * 2);
  _Float16*  gfeat = (_Float16*)alloc((size_t)B_ * 1024 * 2);
  float*     cbuf  = (float*)alloc((size_t)B_ * NPADH * 4);
  float*     nbuf  = (float*)alloc((size_t)B_ * NPADH * 4);
  _Float16*  bufA  = (_Float16*)alloc((size_t)262144 * 128 * 2);   // 64 MB
  _Float16*  bufB  = (_Float16*)alloc((size_t)262144 * 128 * 2);   // 64 MB
  _Float16* w16[13]; float* sc[13]; float* sh[13];
  for (int l = 0; l < 13; ++l) {
    w16[l] = (_Float16*)alloc((size_t)L[l].Npad * L[l].Kpad * 2);
    sc[l]  = (float*)alloc((size_t)L[l].Npad * 4);
    sh[l]  = (float*)alloc((size_t)L[l].Npad * 4);
  }

  auto g1 = [](long n) { return dim3((unsigned)((n + 255) / 256)); };

  // ---- fold BN + convert weights to f16 ----
  for (int l = 0; l < 13; ++l) {
    long nt = (long)L[l].Npad * L[l].Kpad;
    const float* g  = (L[l].ig  >= 0) ? (const float*)d_in[L[l].ig]  : nullptr;
    const float* be = (L[l].ibe >= 0) ? (const float*)d_in[L[l].ibe] : nullptr;
    k_prep<<<g1(nt), 256, 0, stream>>>((const float*)d_in[L[l].iW],
                                       (const float*)d_in[L[l].ib], g, be,
                                       L[l].N, L[l].Kin, L[l].Kpad, L[l].Npad,
                                       w16[l], sc[l], sh[l]);
  }

  // ---- geometry ----
  k_xyzT<<<g1((long)B_ * N_), 256, 0, stream>>>(xyz, ptsT);
  k_fps<<<B_, 256, 0, stream>>>(ptsT, N_, NP1, idx1);
  k_gather_ctr<<<g1((long)B_ * NP1), 256, 0, stream>>>(ptsT, idx1, N_, NP1, ctr1);
  k_ballq<<<B_ * NP1, 32, 0, stream>>>(ptsT, ctr1, N_, NP1, NS1, 0.04f, bq1);

  // ---- SA1: M=262144 rows ----
  k_group1<<<g1((long)B_ * NP1 * NS1), 256, 0, stream>>>(ptsT, ctr1, bq1, bufA);
  k_gemm<1, true><<<dim3(16384, 4),  32, 0, stream>>>(bufA, w16[0],  32, sc[0],  sh[0],  bufB, nullptr, 64,  64);
  k_gemm<1, true><<<dim3(16384, 4),  32, 0, stream>>>(bufB, w16[1],  64, sc[1],  sh[1],  bufA, nullptr, 64,  64);
  k_gemm<1, true><<<dim3(16384, 8),  32, 0, stream>>>(bufA, w16[2],  64, sc[2],  sh[2],  bufB, nullptr, 128, 128);
  k_maxpool<<<g1((long)B_ * NP1 * 128), 256, 0, stream>>>(bufB, l1f, B_ * NP1, NS1, 128);

  // ---- SA2: M=131072 rows ----
  k_fps<<<B_, 256, 0, stream>>>(ctr1, NP1, NP2, idx2);
  k_gather_ctr<<<g1((long)B_ * NP2), 256, 0, stream>>>(ctr1, idx2, NP1, NP2, ctr2);
  k_ballq<<<B_ * NP2, 32, 0, stream>>>(ctr1, ctr2, NP1, NP2, NS2, 0.16f, bq2);
  k_group2<<<g1((long)B_ * NP2 * NS2), 256, 0, stream>>>(ctr1, ctr2, bq2, l1f, bufA);
  k_gemm<1, true><<<dim3(8192, 8),  32, 0, stream>>>(bufA, w16[3], 160, sc[3], sh[3], bufB, nullptr, 128, 128);
  k_gemm<1, true><<<dim3(8192, 8),  32, 0, stream>>>(bufB, w16[4], 128, sc[4], sh[4], bufA, nullptr, 128, 128);
  k_gemm<1, true><<<dim3(8192, 16), 32, 0, stream>>>(bufA, w16[5], 128, sc[5], sh[5], bufB, nullptr, 256, 256);
  k_maxpool<<<g1((long)B_ * NP2 * 256), 256, 0, stream>>>(bufB, l2f, B_ * NP2, NS2, 256);

  // ---- SA3 (group_all): M=2048 rows ----
  k_group3<<<g1((long)B_ * NP2), 256, 0, stream>>>(ctr2, l2f, bufA);
  k_gemm<1, true><<<dim3(128, 16), 32, 0, stream>>>(bufA, w16[6], 288, sc[6], sh[6], bufB, nullptr, 256,  256);
  k_gemm<1, true><<<dim3(128, 32), 32, 0, stream>>>(bufB, w16[7], 256, sc[7], sh[7], bufA, nullptr, 512,  512);
  k_gemm<1, true><<<dim3(128, 64), 32, 0, stream>>>(bufA, w16[8], 512, sc[8], sh[8], bufB, nullptr, 1024, 1024);
  k_maxpool<<<g1((long)B_ * 1024), 256, 0, stream>>>(bufB, gfeat, B_, NP2, 1024);

  // ---- FC head: M=16 ----
  k_gemm<1, true ><<<dim3(1, 64),  32, 0, stream>>>(gfeat, w16[9],  1024, sc[9],  sh[9],  bufA, nullptr, 1024, 1024);
  k_gemm<1, true ><<<dim3(1, 64),  32, 0, stream>>>(bufA,  w16[10], 1024, sc[10], sh[10], bufB, nullptr, 1024, 1024);
  k_gemm<0, false><<<dim3(1, 282), 32, 0, stream>>>(bufB,  w16[11], 1024, sc[11], sh[11], nullptr, cbuf, NPADH, 4500);
  k_gemm<0, false><<<dim3(1, 282), 32, 0, stream>>>(bufB,  w16[12], 1024, sc[12], sh[12], nullptr, nbuf, NPADH, 4500);

  k_final<<<g1((long)B_ * OUTV), 256, 0, stream>>>(cbuf, nbuf, (float*)d_out);
}
